// HungarianMatcher_85160611545672
// MI455X (gfx1250) — matrix operand order repaired and checked by reading, not compiled, not run on hardware
//
#include <hip/hip_runtime.h>
#include <hip/hip_bf16.h>
#include <math.h>

typedef __attribute__((ext_vector_type(16))) _Float16 v16h;
typedef __attribute__((ext_vector_type(8)))  float    v8f;

namespace {
constexpr int kBS = 32;
constexpr int kQ  = 900;
constexpr int kC  = 91;    // num classes
constexpr int kCP = 96;    // padded to multiple of 32 for WMMA K
constexpr int kT  = 30;    // targets per batch
constexpr int kNT = kBS * kT;   // 960 total target columns
constexpr int kNQ = kBS * kQ;   // 28800 total query rows
constexpr int kRowsPerBlk = 16;
}

// ---------------------------------------------------------------------------
// Kernel 1: fused cost matrix.
//   C3[row, j] = 5*L1(box_row, tbox_j) + (1 - P[row, lbl_j]) + 2*(1 - GIoU)
// The class gather P[row, lbl_j] is computed as P @ OneHot via
// v_wmma_f32_16x16x32_f16 (exact: one-hot selection), K = 96 in 3 chunks.
// Block = 256 threads (8 wave32) owns 16 consecutive query rows.
// ---------------------------------------------------------------------------
__global__ __launch_bounds__(256) void hm_cost_kernel(
    const float* __restrict__ logits,   // [28800, 91]
    const float* __restrict__ pboxes,   // [28800, 4]  cxcywh
    const int*   __restrict__ tlabels,  // [960]
    const float* __restrict__ tboxes,   // [960, 4]    cxcywh
    float* __restrict__ out)            // [28800, 960]
{
  __shared__ float sProb[kRowsPerBlk][kCP];  // softmax probs, padded w/ zeros
  __shared__ float sPBox[kRowsPerBlk][4];
  __shared__ float sTBox[kNT][4];
  __shared__ int   sLbl[kNT];

  const int tid  = threadIdx.x;
  const int lane = tid & 31;
  const int wid  = tid >> 5;
  const int row0 = blockIdx.x * kRowsPerBlk;

  // Stage all target data (shared by every row of this block).
  for (int j = tid; j < kNT; j += 256) {
    float4 tb = ((const float4*)tboxes)[j];
    sTBox[j][0] = tb.x; sTBox[j][1] = tb.y;
    sTBox[j][2] = tb.z; sTBox[j][3] = tb.w;
    sLbl[j] = tlabels[j];
  }
  if (tid < kRowsPerBlk) {
    float4 pb = ((const float4*)pboxes)[row0 + tid];
    sPBox[tid][0] = pb.x; sPBox[tid][1] = pb.y;
    sPBox[tid][2] = pb.z; sPBox[tid][3] = pb.w;
  }

  // Softmax: wave w handles rows 2w and 2w+1; lane covers classes
  // {lane, lane+32, lane+64}.
  for (int rr = wid * 2; rr < wid * 2 + 2; ++rr) {
    const int row = row0 + rr;
    const float l0 = logits[(size_t)row * kC + lane];
    const float l1 = logits[(size_t)row * kC + lane + 32];           // 32..63 < 91
    const float l2 = (lane + 64 < kC) ? logits[(size_t)row * kC + lane + 64]
                                      : -INFINITY;
    float m = fmaxf(l0, fmaxf(l1, l2));
    #pragma unroll
    for (int off = 16; off; off >>= 1) m = fmaxf(m, __shfl_xor(m, off, 32));
    const float e0 = __expf(l0 - m);
    const float e1 = __expf(l1 - m);
    const float e2 = (lane + 64 < kC) ? __expf(l2 - m) : 0.0f;
    float s = e0 + e1 + e2;
    #pragma unroll
    for (int off = 16; off; off >>= 1) s += __shfl_xor(s, off, 32);
    const float rs = 1.0f / s;
    sProb[rr][lane]      = e0 * rs;
    sProb[rr][lane + 32] = e1 * rs;
    sProb[rr][lane + 64] = (lane + 64 < kC) ? e2 * rs : 0.0f;  // pad 91..95
  }
  __syncthreads();

  // Build A fragments once (reused for every target tile of this wave).
  // A 16x32 in this code's contiguous k order: lane<16 -> M=lane, K = h;
  // lane>=16 -> M=lane-16, K shifted by +16.
  const int hi   = lane >> 4;       // 0: lanes 0-15, 1: lanes 16-31
  const int msub = lane & 15;
  v16h afrag[3];
  #pragma unroll
  for (int ck = 0; ck < 3; ++ck) {
    #pragma unroll
    for (int h = 0; h < 16; ++h) {
      const int k = h + (hi ? 16 : 0);
      afrag[ck][h] = (_Float16)sProb[msub][ck * 32 + k];
    }
  }

  // Each wave sweeps 16-wide target tiles jt = wid, wid+8, ...
  for (int jt = wid; jt < kNT / 16; jt += 8) {
    const int j0 = jt * 16;
    const int j  = j0 + msub;           // this lane's target column
    const int lbl = sLbl[j];

    // WMMA class gather: acc[r] = P[M, lbl_j] for M = r + 8*hi.
    v8f acc = {0.f, 0.f, 0.f, 0.f, 0.f, 0.f, 0.f, 0.f};
    #pragma unroll
    for (int ck = 0; ck < 3; ++ck) {
      // B 32x16 in the same contiguous k order as A: lane holds column N=lane%16, K = h (+16 for
      // upper lane half). One-hot against this tile's labels.
      const int rel = lbl - ck * 32 - (hi ? 16 : 0);
      v16h bfrag;
      #pragma unroll
      for (int h = 0; h < 16; ++h)
        bfrag[h] = (_Float16)((rel == h) ? 1.0f : 0.0f);
      acc = __builtin_amdgcn_wmma_f32_16x16x32_f16(
          /*neg_a=*/false, afrag[ck], /*neg_b=*/false, bfrag,
          /*c_mod=*/(short)0, acc, /*reuse_a=*/false, /*reuse_b=*/false);
    }

    // Target box (shared across the 8 output rows this lane produces).
    const float tx = sTBox[j][0], ty = sTBox[j][1];
    const float tw = sTBox[j][2], th = sTBox[j][3];
    const float tx0 = tx - 0.5f * tw, ty0 = ty - 0.5f * th;
    const float tx1 = tx + 0.5f * tw, ty1 = ty + 0.5f * th;
    const float areaB = tw * th;

    #pragma unroll
    for (int r = 0; r < 8; ++r) {
      const int M = r + (hi ? 8 : 0);   // C/D layout: vgpr r, lane half -> row
      const float px = sPBox[M][0], py = sPBox[M][1];
      const float pw = sPBox[M][2], ph = sPBox[M][3];
      const float cbbox = fabsf(px - tx) + fabsf(py - ty) +
                          fabsf(pw - tw) + fabsf(ph - th);
      const float px0 = px - 0.5f * pw, py0 = py - 0.5f * ph;
      const float px1 = px + 0.5f * pw, py1 = py + 0.5f * ph;
      const float areaA = pw * ph;
      const float iw = fmaxf(fminf(px1, tx1) - fmaxf(px0, tx0), 0.0f);
      const float ih = fmaxf(fminf(py1, ty1) - fmaxf(py0, ty0), 0.0f);
      const float inter = iw * ih;
      const float uni   = areaA + areaB - inter;
      const float iou   = inter / uni;
      const float ew = fmaxf(px1, tx1) - fminf(px0, tx0);
      const float eh = fmaxf(py1, ty1) - fminf(py0, ty0);
      const float encl = fmaxf(ew, 0.0f) * fmaxf(eh, 0.0f);
      const float giou = iou - (encl - uni) / encl;
      const float cclass = 1.0f - acc[r];
      out[(size_t)(row0 + M) * kNT + j] =
          5.0f * cbbox + 1.0f * cclass + 2.0f * (1.0f - giou);
    }
  }
}

// ---------------------------------------------------------------------------
// Kernel 2: Jonker-Volgenant shortest augmenting path, one workgroup per
// batch. Solves cost.T (n=30 rows x m=900 cols), f64 state as in reference.
// C3 (110 MB) is L2-resident on MI455X (192 MB L2), so strided re-reads of
// cost[i0][:] are cheap.
// ---------------------------------------------------------------------------
__global__ __launch_bounds__(256) void hm_lsa_kernel(
    const float* __restrict__ C,    // [28800, 960]
    float* __restrict__ outPred,    // [32, 30]
    float* __restrict__ outTgt)     // [32, 30]
{
  constexpr int M = kQ;             // 900 columns
  const int b   = blockIdx.x;
  const int tid = threadIdx.x;
  const double INF = 1e18;

  __shared__ double sV[M + 1];
  __shared__ double sMinv[M + 1];
  __shared__ double sU[kT + 1];
  __shared__ int    sWay[M + 1];
  __shared__ int    sP[M + 1];
  __shared__ unsigned char sUsed[M + 1];
  __shared__ double sRedV[256];
  __shared__ int    sRedJ[256];
  __shared__ int    sJ0;
  __shared__ int    sRows[kT];

  // cost[t][q] = C3[b][q][b*kT + t]
  const float* Cb = C + (size_t)b * kQ * kNT + (size_t)b * kT;

  for (int j = tid; j <= M; j += 256) { sV[j] = 0.0; sP[j] = 0; }
  if (tid <= kT) sU[tid] = 0.0;
  __syncthreads();

  for (int i = 1; i <= kT; ++i) {
    if (tid == 0) { sP[0] = i; sJ0 = 0; }
    for (int j = tid; j <= M; j += 256) { sMinv[j] = INF; sUsed[j] = 0; }
    __syncthreads();

    while (true) {
      const int j0 = sJ0;
      if (tid == 0) sUsed[j0] = 1;
      __syncthreads();

      const int    i0  = sP[j0];
      const double ui0 = sU[i0];

      // Update minv over unused columns + local argmin (smallest j on ties,
      // matching np.argmin).
      double bestV = INF; int bestJ = M + 1;
      for (int j = 1 + tid; j <= M; j += 256) {
        if (!sUsed[j]) {
          const double cur =
              (double)Cb[(size_t)(j - 1) * kNT + (i0 - 1)] - ui0 - sV[j];
          double mv = sMinv[j];
          if (cur < mv) { mv = cur; sMinv[j] = cur; sWay[j] = j0; }
          if (mv < bestV || (mv == bestV && j < bestJ)) { bestV = mv; bestJ = j; }
        }
      }
      sRedV[tid] = bestV; sRedJ[tid] = bestJ;
      __syncthreads();
      for (int s = 128; s > 0; s >>= 1) {
        if (tid < s) {
          const double ov = sRedV[tid + s]; const int oj = sRedJ[tid + s];
          if (ov < sRedV[tid] || (ov == sRedV[tid] && oj < sRedJ[tid])) {
            sRedV[tid] = ov; sRedJ[tid] = oj;
          }
        }
        __syncthreads();
      }
      const double delta = sRedV[0];
      const int    j1    = sRedJ[0];

      // Dual updates: used columns map to distinct rows -> no write races.
      for (int j = tid; j <= M; j += 256) {
        if (sUsed[j])     { sU[sP[j]] += delta; sV[j] -= delta; }
        else if (j >= 1)  { sMinv[j] -= delta; }
      }
      if (tid == 0) sJ0 = j1;
      __syncthreads();
      if (sP[j1] == 0) break;     // uniform across block
    }

    if (tid == 0) {               // augment along the stored path
      int j0 = sJ0;
      while (j0) { const int j1 = sWay[j0]; sP[j0] = sP[j1]; j0 = j1; }
    }
    __syncthreads();
  }

  for (int j = 1 + tid; j <= M; j += 256)
    if (sP[j] > 0) sRows[sP[j] - 1] = j - 1;
  __syncthreads();

  // Stable argsort of preds (distinct values -> rank by count of smaller).
  if (tid < kT) {
    const int pr = sRows[tid];
    int rank = 0;
    for (int s = 0; s < kT; ++s) {
      const int ps = sRows[s];
      rank += (ps < pr) || (ps == pr && s < tid);
    }
    outPred[b * kT + rank] = (float)pr;
    outTgt[b * kT + rank]  = (float)tid;
  }
}

extern "C" void kernel_launch(void* const* d_in, const int* in_sizes, int n_in,
                              void* d_out, int out_size, void* d_ws, size_t ws_size,
                              hipStream_t stream) {
  (void)in_sizes; (void)n_in; (void)d_ws; (void)ws_size; (void)out_size;
  const float* logits  = (const float*)d_in[0];  // [32,900,91]
  const float* pboxes  = (const float*)d_in[1];  // [32,900,4]
  const int*   tlabels = (const int*)  d_in[2];  // [32,30]
  const float* tboxes  = (const float*)d_in[3];  // [32,30,4]

  float* out     = (float*)d_out;                      // C3 flat
  float* outPred = out + (size_t)kNQ * kNT;            // + 27,648,000
  float* outTgt  = outPred + (size_t)kBS * kT;         // + 960

  hm_cost_kernel<<<kNQ / kRowsPerBlk, 256, 0, stream>>>(
      logits, pboxes, tlabels, tboxes, out);
  hm_lsa_kernel<<<kBS, 256, 0, stream>>>(out, outPred, outTgt);
}
